// Self_Attention_32384053412301
// MI455X (gfx1250) — compile-verified, matrix operand order checked by reading, not run on hardware
//
#include <hip/hip_runtime.h>
#include <hip/hip_bf16.h>

typedef __attribute__((ext_vector_type(16))) __bf16 v16bf;
typedef __attribute__((ext_vector_type(8)))  float  v8f;

#define WMMA_BF16(A, B, C) \
  __builtin_amdgcn_wmma_f32_16x16x32_bf16(false, (A), false, (B), (short)0, (C), false, false)

__device__ __forceinline__ int iclamp(int x, int lo, int hi) {
  return x < lo ? lo : (x > hi ? hi : x);
}

// Banded self-attention, D=64, band half-width 64.
// One wave (32 lanes) handles one 16-query tile via wave-level flash attention:
//   S^T = K_tile * Q^T     (v_wmma_f32_16x16x32_bf16, K-dim = head dim 64 -> 2 wmma)
//   O  += P * V            (4 wmma per 32-key step, N = 16 of the 64 head dims)
__global__ void __launch_bounds__(128)
banded_attn_wmma(const float* __restrict__ Q,
                 const float* __restrict__ Km,
                 const float* __restrict__ Vm,
                 const int*   __restrict__ mask,
                 float* __restrict__ out,
                 int T) {
  constexpr int D    = 64;
  constexpr int BAND = 64;
  const float NEG_INF = -__builtin_inff();

  const int wave = threadIdx.x >> 5;
  const int lane = threadIdx.x & 31;
  const int n    = lane & 15;   // query index within tile (S^T / A-matrix M / B-matrix N)
  const int half = lane >> 4;   // lane half selects K-subset in 16-bit A/B layouts
  const int hbase = half * 8;

  const int b  = blockIdx.y;
  const int i0 = (blockIdx.x * 4 + wave) * 16;   // first query row of this tile
  const int i  = i0 + n;                          // this lane's query row

  const float* Qb = Q  + (size_t)b * T * D;
  const float* Kb = Km + (size_t)b * T * D;
  const float* Vb = Vm + (size_t)b * T * D;
  const int*  mrow = mask + (size_t)b * T;

  // ---- Load Q^T as B-matrix (bf16 16x32 layout), split over d=[0,32) and [32,64) ----
  // 16-bit B layout: lane column = lane&15; halfword h=2v+e holds
  //   K(v,e,half) = (v<4?0:16) + 2*(v&3) + e + 8*half
  const float* qrow = Qb + (size_t)i * D;
  v16bf qB0, qB1;
#pragma unroll
  for (int v = 0; v < 8; ++v) {
    const int base = (v < 4) ? 0 : 16;
#pragma unroll
    for (int e = 0; e < 2; ++e) {
      const int d = base + 2 * (v & 3) + e + 8 * half;
      qB0[2 * v + e] = (__bf16)qrow[d];
      qB1[2 * v + e] = (__bf16)qrow[d + 32];
    }
  }
  const int mi = mrow[i];  // query-row mask bit for this lane

  // Online-softmax state (per query column; mirrored across lane pair {n, n+16})
  float m_run = NEG_INF;
  float l_run = 0.0f;
  v8f o[4] = {v8f{}, v8f{}, v8f{}, v8f{}};  // O accumulators, d-groups of 16

  // Score one 16-key tile: S^T = K_tile * Q^T, then scale + band/mask -> sarr[8]
  auto score_tile = [&](int jb, float* sarr) {
    const int jr = iclamp(jb + n, 0, T - 1);     // A-matrix row = lane&15
    const float* krow = Kb + (size_t)jr * D;
    v16bf kA0, kA1;
#pragma unroll
    for (int v = 0; v < 8; ++v) {
      const int base = (v < 4) ? 0 : 16;
#pragma unroll
      for (int e = 0; e < 2; ++e) {
        const int d = base + 2 * (v & 3) + e + 8 * half;
        kA0[2 * v + e] = (__bf16)krow[d];
        kA1[2 * v + e] = (__bf16)krow[d + 32];
      }
    }
    v8f c = {};
    c = WMMA_BF16(kA0, qB0, c);   // d = 0..31
    c = WMMA_BF16(kA1, qB1, c);   // d = 32..63
#pragma unroll
    for (int r = 0; r < 8; ++r) {
      const int j   = jb + hbase + r;            // key row (C layout: M = r + 8*half)
      const bool inr = (j >= 0) && (j < T);
      const int  mj  = inr ? mrow[j] : 0;
      const int  dd  = i - j;
      const int  ad  = dd < 0 ? -dd : dd;
      const bool ok  = inr && (mj != 0) && ((ad <= BAND) || (mi == 0));
      sarr[r] = ok ? (c[r] * 0.125f) : NEG_INF;  // 1/sqrt(64)
    }
  };

  // Key window [i0-80, i0+80): covers band [i0-64, i0+79]; 5 steps x 32 keys
#pragma unroll 1
  for (int s = 0; s < 5; ++s) {
    const int jblk = i0 - 80 + 32 * s;

    float s0a[8], s1a[8];
    score_tile(jblk,      s0a);
    score_tile(jblk + 16, s1a);

    // --- online softmax update (per query; merge lane halves) ---
    float mx = NEG_INF;
#pragma unroll
    for (int r = 0; r < 8; ++r) mx = fmaxf(mx, fmaxf(s0a[r], s1a[r]));
    mx = fmaxf(mx, __shfl_xor(mx, 16, 32));
    const float mn = fmaxf(m_run, mx);
    const float alpha = (mn == m_run) ? 1.0f : __expf(m_run - mn);
    m_run = mn;

    float p0[8], p1[8];
    float ls = 0.0f;
#pragma unroll
    for (int r = 0; r < 8; ++r) {
      p0[r] = (s0a[r] == NEG_INF) ? 0.0f : __expf(s0a[r] - mn);
      p1[r] = (s1a[r] == NEG_INF) ? 0.0f : __expf(s1a[r] - mn);
      ls += p0[r] + p1[r];
    }
    ls += __shfl_xor(ls, 16, 32);
    l_run = l_run * alpha + ls;

    // Rescale O accumulators: need alpha for query q = r + 8*half (O layout)
    float aq[8];
#pragma unroll
    for (int r = 0; r < 8; ++r) aq[r] = __shfl(alpha, hbase + r, 32);
#pragma unroll
    for (int g = 0; g < 4; ++g)
#pragma unroll
      for (int r = 0; r < 8; ++r) o[g][r] *= aq[r];

    // Pack P into bf16 A-matrix: keys this lane holds from the two S^T tiles land
    // exactly in this lane's A K-slots (no cross-lane traffic).
    v16bf pA;
#pragma unroll
    for (int v = 0; v < 8; ++v) {
#pragma unroll
      for (int e = 0; e < 2; ++e) {
        const int r = 2 * (v & 3) + e;
        pA[2 * v + e] = (__bf16)((v < 4) ? p0[r] : p1[r]);
      }
    }

    // O += P * V over this 32-key block, 16 head-dims per wmma
#pragma unroll
    for (int g = 0; g < 4; ++g) {
      v16bf vB;
#pragma unroll
      for (int v = 0; v < 8; ++v) {
        const int base = (v < 4) ? 0 : 16;
#pragma unroll
        for (int e = 0; e < 2; ++e) {
          const int kk = base + 2 * (v & 3) + e + 8 * half;   // key within block
          const int jc = iclamp(jblk + kk, 0, T - 1);          // masked keys have p==0
          vB[2 * v + e] = (__bf16)Vb[(size_t)jc * D + 16 * g + n];
        }
      }
      o[g] = WMMA_BF16(pA, vB, o[g]);
    }
  }

  // --- normalize and store: O layout lane N = d-col (within group), row M = r+8*half ---
  float lq[8];
#pragma unroll
  for (int r = 0; r < 8; ++r) lq[r] = __shfl(l_run, hbase + r, 32);
#pragma unroll
  for (int r = 0; r < 8; ++r) {
    const float rcp = (lq[r] > 0.0f) ? (1.0f / lq[r]) : 0.0f;
    const int row = i0 + hbase + r;
#pragma unroll
    for (int g = 0; g < 4; ++g) {
      out[((size_t)b * T + row) * D + 16 * g + n] = o[g][r] * rcp;
    }
  }
}

extern "C" void kernel_launch(void* const* d_in, const int* in_sizes, int n_in,
                              void* d_out, int out_size, void* d_ws, size_t ws_size,
                              hipStream_t stream) {
  const float* Q    = (const float*)d_in[0];
  const float* K    = (const float*)d_in[1];
  const float* V    = (const float*)d_in[2];
  const int*   mask = (const int*)d_in[3];
  float* out = (float*)d_out;

  const int B = 4;
  const int T = in_sizes[3] / B;   // mask is (B, T)
  const int qtiles = T / 16;       // 16-query tiles per batch

  dim3 block(128);                 // 4 waves, one 16-query tile each
  dim3 grid(qtiles / 4, B);
  banded_attn_wmma<<<grid, block, 0, stream>>>(Q, K, V, mask, out, T);
}
